// _SelfAttentionBlock_3865470567062
// MI455X (gfx1250) — matrix operand order repaired and checked by reading, not compiled, not run on hardware
//
#include <hip/hip_runtime.h>
#include <hip/hip_bf16.h>

// ---------------------------------------------------------------------------
// CDNA5 (gfx1250) implementation of the SelfAttentionBlock reference.
// All matmul-like work (1x1 convs, implicit-GEMM KxK convs, QK^T, PV) runs
// through V_WMMA_F32_16X16X32_BF16 (bf16 operands, fp32 accum). Attention is
// flash-style (no materialized 4096x4096 score matrix).
// ---------------------------------------------------------------------------

typedef __attribute__((ext_vector_type(16))) __bf16 v16bf;
typedef __attribute__((ext_vector_type(8)))  __bf16 v8bf;
typedef __attribute__((ext_vector_type(8)))  float  v8f;

#define BN_EPS 1e-5f

// ---------------------------------------------------------------------------
// Generic conv as implicit GEMM:  M = Cout, Ncols = N*OH*OW, K = Cin*KH*KW.
// Workgroup = 128 threads (4 waves). WG tile = 32 (M) x 64 (cols); each wave
// owns 16 columns and computes BOTH 16-row M-halves from one B fragment
// (2 wmma per staged B tile -> 2x arithmetic intensity on the im2col gather).
// Layouts: in_nchw/out_nchw==0 means internal channel-major [C][N*H*W].
// ---------------------------------------------------------------------------
struct ConvP {
  const float* x; const float* w; const float* bias; float* y;
  int N, Cin, H, W, Cout, OH, OW, KH, KW, stride, padH, padW, dil;
  int in_nchw, out_nchw;
};

__global__ __launch_bounds__(128) void conv_wmma(ConvP p) {
  __shared__ __align__(32) __bf16 As[32 * 32];   // [m][k]
  __shared__ __align__(32) __bf16 Bs[64 * 32];   // [col][k] (transposed for frag reads)
  const int tid  = threadIdx.x;
  const int wave = tid >> 5;
  const int lane = tid & 31;
  const int mbase   = blockIdx.y * 32;
  const int colbase = blockIdx.x * 64;
  const int ohw   = p.OH * p.OW;
  const int Ncols = p.N * ohw;
  const int taps  = p.KH * p.KW;
  const long inNtot = (long)p.N * p.H * p.W;

  v8f acc0 = {}, acc1 = {};

  for (int kh = 0; kh < p.KH; ++kh)
  for (int kw = 0; kw < p.KW; ++kw)
  for (int ci0 = 0; ci0 < p.Cin; ci0 += 32) {
    // --- stage A tile (32 x 32 weights) ---
    for (int i = tid; i < 1024; i += 128) {
      int r = i >> 5, kk = i & 31;
      int m = mbase + r, ci = ci0 + kk;
      float v = 0.f;
      if (m < p.Cout && ci < p.Cin)
        v = p.w[((long)m * p.Cin + ci) * taps + kh * p.KW + kw];
      As[r * 32 + kk] = (__bf16)v;
    }
    // --- stage B tile (32 x 64 im2col gather), coalesced along output col ---
    for (int i = tid; i < 2048; i += 128) {
      int kk = i >> 6, col = i & 63;
      int cg = colbase + col;
      int ci = ci0 + kk;
      float v = 0.f;
      if (cg < Ncols && ci < p.Cin) {
        int n = cg / ohw, rem = cg - n * ohw;
        int oh = rem / p.OW, ow = rem - oh * p.OW;
        int ih = oh * p.stride - p.padH + kh * p.dil;
        int iw = ow * p.stride - p.padW + kw * p.dil;
        if (ih >= 0 && ih < p.H && iw >= 0 && iw < p.W) {
          long idx = p.in_nchw
              ? (((long)(n * p.Cin + ci) * p.H + ih) * p.W + iw)
              : ((long)ci * inNtot + ((long)n * p.H + ih) * p.W + iw);
          v = p.x[idx];
        }
      }
      Bs[col * 32 + kk] = (__bf16)v;
    }
    __syncthreads();

    // --- build fragments per ISA 7.12.2 layouts ---
    const int mm = lane & 15;
    const int half = (lane >> 4) & 1;
    const int b0 = half * 8;
    v16bf a0, a1, b;
    *reinterpret_cast<v8bf*>(&a0)       = *reinterpret_cast<const v8bf*>(&As[mm * 32 + b0]);
    *(reinterpret_cast<v8bf*>(&a0) + 1) = *reinterpret_cast<const v8bf*>(&As[mm * 32 + b0 + 16]);
    *reinterpret_cast<v8bf*>(&a1)       = *reinterpret_cast<const v8bf*>(&As[(16 + mm) * 32 + b0]);
    *(reinterpret_cast<v8bf*>(&a1) + 1) = *reinterpret_cast<const v8bf*>(&As[(16 + mm) * 32 + b0 + 16]);
    const int colw = wave * 16 + mm;
    *reinterpret_cast<v8bf*>(&b)       = *reinterpret_cast<const v8bf*>(&Bs[colw * 32 + b0]);
    *(reinterpret_cast<v8bf*>(&b) + 1) = *reinterpret_cast<const v8bf*>(&Bs[colw * 32 + b0 + 16]);

    acc0 = __builtin_amdgcn_wmma_f32_16x16x32_bf16(false, a0, false, b, (short)0, acc0,
                                                   false, false);
    acc1 = __builtin_amdgcn_wmma_f32_16x16x32_bf16(false, a1, false, b, (short)0, acc1,
                                                   false, false);
    __syncthreads();
  }

  // --- store C tiles (VGPR i -> row i/(8+i) per lane-half, col = lane%16) ---
  const int half = (lane >> 4) & 1;
  const int col = colbase + wave * 16 + (lane & 15);
  if (col < Ncols) {
    int n = col / ohw, pos = col - n * ohw;
    #pragma unroll
    for (int i = 0; i < 8; ++i) {
      int m0 = mbase + i + half * 8;
      if (m0 < p.Cout) {
        float v = acc0[i] + (p.bias ? p.bias[m0] : 0.f);
        long oidx = p.out_nchw ? ((long)(n * p.Cout + m0) * ohw + pos)
                               : ((long)m0 * Ncols + col);
        p.y[oidx] = v;
      }
      int m1 = m0 + 16;
      if (m1 < p.Cout) {
        float v = acc1[i] + (p.bias ? p.bias[m1] : 0.f);
        long oidx = p.out_nchw ? ((long)(n * p.Cout + m1) * ohw + pos)
                               : ((long)m1 * Ncols + col);
        p.y[oidx] = v;
      }
    }
  }
}

// ---------------------------------------------------------------------------
// Training-mode BatchNorm: per-channel mean/var over the contiguous
// [C][N*H*W] row, folded into scale/shift; then an elementwise apply pass.
// ---------------------------------------------------------------------------
__global__ __launch_bounds__(256) void bn_stats(const float* x, const float* g,
                                                const float* beta, float* ss, int Ntot) {
  __shared__ float s1[256], s2[256];
  const int c = blockIdx.x;
  const float* row = x + (long)c * Ntot;
  float a = 0.f, b = 0.f;
  for (int i = threadIdx.x; i < Ntot; i += 256) { float v = row[i]; a += v; b += v * v; }
  s1[threadIdx.x] = a; s2[threadIdx.x] = b;
  __syncthreads();
  for (int s = 128; s > 0; s >>= 1) {
    if (threadIdx.x < s) { s1[threadIdx.x] += s1[threadIdx.x + s];
                           s2[threadIdx.x] += s2[threadIdx.x + s]; }
    __syncthreads();
  }
  if (threadIdx.x == 0) {
    float mean = s1[0] / (float)Ntot;
    float var  = s2[0] / (float)Ntot - mean * mean;
    float sc   = g[c] * rsqrtf(var + BN_EPS);
    ss[c]       = sc;
    ss[c + 256] = beta[c] - mean * sc;
  }
}

__global__ void bn_apply(float* y, const float* ss, int Ntot, int relu) {
  const int c = blockIdx.y;
  const int i = blockIdx.x * 256 + threadIdx.x;
  if (i < Ntot) {
    long idx = (long)c * Ntot + i;
    float v = y[idx] * ss[c] + ss[c + 256];
    y[idx] = relu ? fmaxf(v, 0.f) : v;
  }
}

__global__ void residual_relu(const float* lin, const float* x, float* out, long n) {
  long i = (long)blockIdx.x * 256 + threadIdx.x;
  if (i < n) out[i] = fmaxf(lin[i] * 0.1f + x[i], 0.f);
}

// ---------------------------------------------------------------------------
// bf16 packing for attention operands.
// pack_rowmajor: [C][N*L] fp32 -> [n][row][C] bf16   (Q, K)
// pack_ct:       [C][N*L] fp32 -> [n][C][L]  bf16    (V^T)
// ---------------------------------------------------------------------------
__global__ void pack_rowmajor_bf16(const float* x, __bf16* y, int Ntot) {
  long i = (long)blockIdx.x * 256 + threadIdx.x;
  if (i < (long)Ntot * 128) {
    int c = (int)(i & 127);
    long t = i >> 7;
    y[i] = (__bf16)x[(long)c * Ntot + t];
  }
}

__global__ void pack_ct_bf16(const float* x, __bf16* y) {
  long i = (long)blockIdx.x * 256 + threadIdx.x;
  if (i < (long)8 * 128 * 4096) {
    int j = (int)(i & 4095);
    long w = i >> 12;
    int c = (int)(w & 127);
    int n = (int)(w >> 7);
    y[i] = (__bf16)x[(long)c * 32768 + (long)n * 4096 + j];
  }
}

// ---------------------------------------------------------------------------
// Flash attention. Lq = Lk = 4096, c = 128, scale = 1/sqrt(128).
// Grid (64, 8): 64 query-tiles of 64 per batch; 4 waves/WG, 16 queries/wave.
// Per wave: Q resident (4 A-frags), O = 8 f32 C-tiles, online softmax with
// 16-lane butterflies, P re-laid out C->A through per-wave LDS.
// K/V stream prefetched one 32-key tile ahead (global_prefetch_b8).
// ---------------------------------------------------------------------------
__global__ __launch_bounds__(128) void flash_attn(const __bf16* Qb, const __bf16* Kb,
                                                  const __bf16* Vt, float* ctx) {
  __shared__ __align__(32) __bf16 Ps[4][16 * 32];
  const int n = blockIdx.y;
  const int wave = threadIdx.x >> 5;
  const int lane = threadIdx.x & 31;
  const int qbase = blockIdx.x * 64 + wave * 16;
  const __bf16* Qn = Qb + (long)n * 4096 * 128;
  const __bf16* Kn = Kb + (long)n * 4096 * 128;
  const __bf16* Vn = Vt + (long)n * 128 * 4096;
  const int mm = lane & 15;
  const int half = (lane >> 4) & 1;
  const int b0 = half * 8;
  const float scale = 0.08838834764831845f;  // 128^-0.5

  // Q fragments (held for the whole K loop)
  v16bf qf[4];
  #pragma unroll
  for (int cc = 0; cc < 4; ++cc) {
    const __bf16* qp = Qn + (long)(qbase + mm) * 128 + cc * 32 + b0;
    *reinterpret_cast<v8bf*>(&qf[cc])       = *reinterpret_cast<const v8bf*>(qp);
    *(reinterpret_cast<v8bf*>(&qf[cc]) + 1) = *reinterpret_cast<const v8bf*>(qp + 16);
  }

  v8f o[8] = {};
  float mrow[8], lrow[8];
  #pragma unroll
  for (int i = 0; i < 8; ++i) { mrow[i] = -3.0e38f; lrow[i] = 0.f; }

  for (int jb = 0; jb < 4096; jb += 32) {
    // prefetch next key tile (K rows 256B each; V rows: 64B span per tile)
    if (jb + 32 < 4096) {
      __builtin_prefetch(Kn + (long)(jb + 32 + lane) * 128, 0, 1);
      __builtin_prefetch(Vn + (long)(lane * 4) * 4096 + jb + 32, 0, 1);
    }

    // S = Q * K^T for 32 keys (two 16-col C tiles), K-loop over c in 4 chunks
    v8f s0 = {}, s1 = {};
    #pragma unroll
    for (int cc = 0; cc < 4; ++cc) {
      v16bf k0, k1;
      *reinterpret_cast<v8bf*>(&k0)       = *reinterpret_cast<const v8bf*>(Kn + (long)(jb + mm) * 128 + cc * 32 + b0);
      *(reinterpret_cast<v8bf*>(&k0) + 1) = *reinterpret_cast<const v8bf*>(Kn + (long)(jb + mm) * 128 + cc * 32 + b0 + 16);
      *reinterpret_cast<v8bf*>(&k1)       = *reinterpret_cast<const v8bf*>(Kn + (long)(jb + 16 + mm) * 128 + cc * 32 + b0);
      *(reinterpret_cast<v8bf*>(&k1) + 1) = *reinterpret_cast<const v8bf*>(Kn + (long)(jb + 16 + mm) * 128 + cc * 32 + b0 + 16);
      s0 = __builtin_amdgcn_wmma_f32_16x16x32_bf16(false, qf[cc], false, k0, (short)0, s0, false, false);
      s1 = __builtin_amdgcn_wmma_f32_16x16x32_bf16(false, qf[cc], false, k1, (short)0, s1, false, false);
    }

    // online softmax; row r = i + half*8, col = mm (C layout)
    #pragma unroll
    for (int i = 0; i < 8; ++i) {
      float a = s0[i] * scale, b = s1[i] * scale;
      float mx = fmaxf(a, b);
      mx = fmaxf(mx, __shfl_xor(mx, 1));
      mx = fmaxf(mx, __shfl_xor(mx, 2));
      mx = fmaxf(mx, __shfl_xor(mx, 4));
      mx = fmaxf(mx, __shfl_xor(mx, 8));
      float mnew  = fmaxf(mrow[i], mx);
      float alpha = __expf(mrow[i] - mnew);
      float p0 = __expf(a - mnew);
      float p1 = __expf(b - mnew);
      float rs = p0 + p1;
      rs += __shfl_xor(rs, 1);
      rs += __shfl_xor(rs, 2);
      rs += __shfl_xor(rs, 4);
      rs += __shfl_xor(rs, 8);
      lrow[i] = lrow[i] * alpha + rs;
      mrow[i] = mnew;
      #pragma unroll
      for (int cc = 0; cc < 8; ++cc) o[cc][i] *= alpha;
      int r = i + half * 8;
      Ps[wave][r * 32 + mm]      = (__bf16)p0;
      Ps[wave][r * 32 + 16 + mm] = (__bf16)p1;
    }

    // same-wave LDS RAW: DS is in-order per wave; wait + compiler barrier
    asm volatile("s_wait_dscnt 0" ::: "memory");

    // P as A-frag (16 q x 32 j), V as B-frags (32 j x 16 c) from Vt[n][c][j]
    v16bf pf;
    const __bf16* pp = &Ps[wave][mm * 32 + b0];
    *reinterpret_cast<v8bf*>(&pf)       = *reinterpret_cast<const v8bf*>(pp);
    *(reinterpret_cast<v8bf*>(&pf) + 1) = *reinterpret_cast<const v8bf*>(pp + 16);
    #pragma unroll
    for (int cc = 0; cc < 8; ++cc) {
      v16bf vf;
      *reinterpret_cast<v8bf*>(&vf)       = *reinterpret_cast<const v8bf*>(Vn + (long)(cc * 16 + mm) * 4096 + jb + b0);
      *(reinterpret_cast<v8bf*>(&vf) + 1) = *reinterpret_cast<const v8bf*>(Vn + (long)(cc * 16 + mm) * 4096 + jb + b0 + 16);
      o[cc] = __builtin_amdgcn_wmma_f32_16x16x32_bf16(false, pf, false, vf, (short)0, o[cc], false, false);
    }
  }

  // epilogue: ctx stored channel-major [c][n*4096 + q] for the final 1x1 conv
  #pragma unroll
  for (int i = 0; i < 8; ++i) {
    float rl = 1.f / lrow[i];
    int q = qbase + i + half * 8;
    #pragma unroll
    for (int cc = 0; cc < 8; ++cc) {
      int c = cc * 16 + mm;
      ctx[(long)c * 32768 + (long)n * 4096 + q] = o[cc][i] * rl;
    }
  }
}

// ---------------------------------------------------------------------------
// Host orchestration.
// ---------------------------------------------------------------------------
extern "C" void kernel_launch(void* const* d_in, const int* in_sizes, int n_in,
                              void* d_out, int out_size, void* d_ws, size_t ws_size,
                              hipStream_t stream) {
  (void)in_sizes; (void)n_in; (void)out_size; (void)ws_size;
  const int N = 8;
  const int L1 = 128 * 128, L2 = 64 * 64;
  const int NT1 = N * L1;            // 131072
  const int NT2 = N * L2;            // 32768

  auto F = [&](int i) { return (const float*)d_in[i]; };

  // ---- workspace layout (sequentially reused regions) ----
  char* ws = (char*)d_ws;
  size_t off = 0;
  auto alloc = [&](size_t bytes) {
    size_t o = off; off += (bytes + 255) & ~(size_t)255; return o;
  };
  float* v0   = (float*)(ws + alloc((size_t)128 * NT1 * 4));  // value 1x1 out / RFB x
  float* tA   = (float*)(ws + alloc((size_t)32  * NT1 * 4));  // RFB scratch
  float* tB   = (float*)(ws + alloc((size_t)32  * NT1 * 4));  // RFB scratch
  float* cat  = (float*)(ws + alloc((size_t)128 * NT1 * 4));  // RFB concat/out
  float* lin  = (float*)(ws + alloc((size_t)128 * NT1 * 4));  // RFB lin out; later ctx
  float* v1   = (float*)(ws + alloc((size_t)128 * NT2 * 4));  // 64x64 stage buffer
  float* stats= (float*)(ws + alloc(512 * 4));                // BN scale/shift
  __bf16* Qb  = (__bf16*)(ws + alloc((size_t)NT2 * 128 * 2));
  __bf16* Kb  = (__bf16*)(ws + alloc((size_t)NT2 * 128 * 2));
  __bf16* Vt  = (__bf16*)(ws + alloc((size_t)NT2 * 128 * 2));
  float* ctx  = lin;  // lin region is free by attention time

  auto conv = [&](const float* x, const float* w, const float* bias, float* y,
                  int Cin, int Cout, int H, int W, int KH, int KW,
                  int s, int ph, int pw, int dil, int in_nchw, int out_nchw) {
    int OH = (H + 2 * ph - dil * (KH - 1) - 1) / s + 1;
    int OW = (W + 2 * pw - dil * (KW - 1) - 1) / s + 1;
    ConvP p{x, w, bias, y, N, Cin, H, W, Cout, OH, OW, KH, KW, s, ph, pw, dil,
            in_nchw, out_nchw};
    int Ncols = N * OH * OW;
    dim3 g((Ncols + 63) / 64, (Cout + 31) / 32);
    conv_wmma<<<g, 128, 0, stream>>>(p);
  };
  auto bnrelu = [&](float* y, const float* g_, const float* b_, int C, int Ntot, int relu) {
    bn_stats<<<C, 256, 0, stream>>>(y, g_, b_, stats, Ntot);
    bn_apply<<<dim3((Ntot + 255) / 256, C), 256, 0, stream>>>(y, stats, Ntot, relu);
  };
  // BasicConv: conv(bias=False) + BN(train) + optional ReLU. wi -> {w,g,beta}
  auto bc = [&](const float* x, int wi, int Cin, int Cout, int H, int W,
                int KH, int KW, int s, int ph, int pw, int dil, float* y, int relu) {
    conv(x, F(wi), nullptr, y, Cin, Cout, H, W, KH, KW, s, ph, pw, dil, 0, 0);
    int OH = (H + 2 * ph - dil * (KH - 1) - 1) / s + 1;
    int OW = (W + 2 * pw - dil * (KW - 1) - 1) / s + 1;
    bnrelu(y, F(wi + 1), F(wi + 2), Cout, N * OH * OW, relu);
  };
  // BasicRFB_a(128->128); branch outputs land at channel offsets of `cat`.
  auto rfb = [&](const float* x, int H, int W) {
    long NT = (long)N * H * W;
    bc(x,  19, 128, 32, H, W, 1, 1, 1, 0, 0, 1, tA, 1);
    bc(tA, 22,  32, 32, H, W, 3, 3, 1, 1, 1, 1, cat + 0 * NT, 0);
    bc(x,  25, 128, 32, H, W, 1, 1, 1, 0, 0, 1, tA, 1);
    bc(tA, 28,  32, 32, H, W, 3, 1, 1, 1, 0, 1, tB, 1);
    bc(tB, 31,  32, 32, H, W, 3, 3, 1, 3, 3, 3, cat + 32 * NT, 0);
    bc(x,  34, 128, 32, H, W, 1, 1, 1, 0, 0, 1, tA, 1);
    bc(tA, 37,  32, 32, H, W, 1, 3, 1, 0, 1, 1, tB, 1);
    bc(tB, 40,  32, 32, H, W, 3, 3, 1, 3, 3, 3, cat + 64 * NT, 0);
    bc(x,  43, 128, 16, H, W, 1, 1, 1, 0, 0, 1, tA, 1);
    bc(tA, 46,  16, 24, H, W, 1, 3, 1, 0, 1, 1, tB, 1);
    bc(tB, 49,  24, 32, H, W, 3, 1, 1, 1, 0, 1, tA, 1);
    bc(tA, 52,  32, 32, H, W, 3, 3, 1, 5, 5, 5, cat + 96 * NT, 0);
    bc(cat, 55, 128, 128, H, W, 1, 1, 1, 0, 0, 1, lin, 0);
    long tot = 128 * NT;
    residual_relu<<<(unsigned)((tot + 255) / 256), 256, 0, stream>>>(lin, x, cat, tot);
  };

  // ===== value path: 1x1(bias) -> RFB -> 3x3 s2 (bias) -> BN -> pack V^T =====
  conv(F(0), F(11), F(12), v0, 256, 128, 128, 128, 1, 1, 1, 0, 0, 1, 1, 0);
  rfb(v0, 128, 128);                               // -> cat [128][NT1]
  conv(cat, F(15), F(16), v1, 128, 128, 128, 128, 3, 3, 2, 1, 1, 1, 0, 0);
  bnrelu(v1, F(17), F(18), 128, NT2, 0);
  {
    long tot = (long)128 * NT2;
    pack_ct_bf16<<<(unsigned)((tot + 255) / 256), 256, 0, stream>>>(v1, Vt);
  }

  // ===== query path: 1x1(bias) -> BN -> ReLU -> pack Q =====
  conv(F(2), F(7), F(8), v1, 128, 128, 64, 64, 1, 1, 1, 0, 0, 1, 1, 0);
  bnrelu(v1, F(9), F(10), 128, NT2, 1);
  {
    long tot = (long)128 * NT2;
    pack_rowmajor_bf16<<<(unsigned)((tot + 255) / 256), 256, 0, stream>>>(v1, Qb, NT2);
  }

  // ===== key path: 1x1(bias) -> BN -> ReLU -> RFB -> pack K =====
  conv(F(1), F(3), F(4), v1, 256, 128, 64, 64, 1, 1, 1, 0, 0, 1, 1, 0);
  bnrelu(v1, F(5), F(6), 128, NT2, 1);
  rfb(v1, 64, 64);                                 // -> cat [128][NT2]
  {
    long tot = (long)128 * NT2;
    pack_rowmajor_bf16<<<(unsigned)((tot + 255) / 256), 256, 0, stream>>>(cat, Kb, NT2);
  }

  // ===== attention (flash) -> ctx [c][n*4096+q] =====
  flash_attn<<<dim3(64, 8), 128, 0, stream>>>(Qb, Kb, Vt, ctx);

  // ===== final 1x1 conv W -> d_out (NCHW) =====
  conv(ctx, F(13), F(14), (float*)d_out, 128, 128, 64, 64, 1, 1, 1, 0, 0, 1, 0, 1);
}
